// DreamAttention_48155173322898
// MI455X (gfx1250) — matrix operand order repaired and checked by reading, not compiled, not run on hardware
//
#include <hip/hip_runtime.h>

#define T_SEQ 2048
#define HID   3584
#define NH    28
#define NKV   4
#define HD    128
#define GQA   (NH / NKV)
#define KVW   (NKV * HD)
#define ATT_SCALE 0.08838834764831845f   // 128^-0.5

typedef __attribute__((ext_vector_type(16))) __bf16 bf16x16;
typedef __attribute__((ext_vector_type(8)))  float  f32x8;

union Frag16 { bf16x16 v; unsigned u[8]; uint4 q[2]; };

__device__ __forceinline__ unsigned short f2bf(float f) {
  unsigned u = __float_as_uint(f);
  u += 0x7fffu + ((u >> 16) & 1u);           // round-to-nearest-even
  return (unsigned short)(u >> 16);
}

__device__ __forceinline__ f32x8 zero8() {
  f32x8 z = {0.f, 0.f, 0.f, 0.f, 0.f, 0.f, 0.f, 0.f};
  return z;
}

__device__ __forceinline__ f32x8 wmma_bf16(const bf16x16& a, const bf16x16& b, const f32x8& c) {
  // D = A(16x32 bf16) * B(32x16 bf16) + C(16x16 f32)
  return __builtin_amdgcn_wmma_f32_16x16x32_bf16(false, a, false, b, (short)0, c, false, false);
}

// A-fragment (16x32 bf16, K contiguous in memory): two b128 loads.
// Lane l16 = row; half-select hs picks K 0-7/16-23 vs 8-15/24-31.
__device__ __forceinline__ Frag16 load_afrag(const unsigned short* row, int hs) {
  Frag16 f;
  f.q[0] = *(const uint4*)(row + hs * 8);
  f.q[1] = *(const uint4*)(row + 16 + hs * 8);
  return f;
}

// B-fragment (32x16 bf16, K contiguous in memory per column row): two b128 loads.
// Lane l16 = column; hs picks K 0-7/16-23 vs 8-15/24-31.
__device__ __forceinline__ Frag16 load_bfrag(const unsigned short* colrow, int hs) {
  Frag16 f;
  f.q[0] = *(const uint4*)(colrow + hs * 8);
  f.q[1] = *(const uint4*)(colrow + 16 + hs * 8);
  return f;
}

// ---------------------------------------------------------------------------
// Register-only TN GEMM: C(MxN,f32) = A(MxK bf16 row-major) @ BT(NxK bf16)^T
// + bias. Block 128x256, 8 waves (2M x 4N), wave tile 64x64, K step 32.
// No LDS, no barriers: all fragments are direct global b128 loads (L0/L2 hit).
// ---------------------------------------------------------------------------
__global__ __launch_bounds__(256)
void gemm_tn_bf16(const unsigned short* __restrict__ A,
                  const unsigned short* __restrict__ BT,
                  const float* __restrict__ bias, float* __restrict__ C,
                  int N, int K)
{
  const int tid  = threadIdx.x;
  const int lane = tid & 31;
  const int wave = tid >> 5;
  const int l16  = lane & 15;
  const int hs   = (lane >> 4) & 1;
  const int bm   = blockIdx.x * 128;
  const int bn   = blockIdx.y * 256;
  const int wM   = (wave & 1) * 64;
  const int wN   = (wave >> 1) * 64;

  f32x8 acc[4][4];
#pragma unroll
  for (int i = 0; i < 4; ++i)
#pragma unroll
    for (int j = 0; j < 4; ++j) acc[i][j] = zero8();

  const unsigned short* aRow = A  + (size_t)(bm + wM + l16) * K;
  const unsigned short* bRow = BT + (size_t)(bn + wN + l16) * K;

  for (int k0 = 0; k0 < K; k0 += 32) {
    // WGP-scope prefetch (locality 3 -> near scope): pull next panels into WGP$
    __builtin_prefetch(aRow + k0 + 128, 0, 3);
    __builtin_prefetch(bRow + k0 + 128, 0, 3);
    Frag16 af[4], bfr[4];
#pragma unroll
    for (int mt = 0; mt < 4; ++mt)
      af[mt] = load_afrag(aRow + (size_t)mt * 16 * K + k0, hs);
#pragma unroll
    for (int nt = 0; nt < 4; ++nt)
      bfr[nt] = load_bfrag(bRow + (size_t)nt * 16 * K + k0, hs);
#pragma unroll
    for (int mt = 0; mt < 4; ++mt)
#pragma unroll
      for (int nt = 0; nt < 4; ++nt)
        acc[mt][nt] = wmma_bf16(af[mt].v, bfr[nt].v, acc[mt][nt]);
  }

#pragma unroll
  for (int mt = 0; mt < 4; ++mt)
#pragma unroll
    for (int nt = 0; nt < 4; ++nt)
#pragma unroll
      for (int v = 0; v < 8; ++v) {
        int row = bm + wM + mt*16 + v + hs*8;            // C layout
        int col = bn + wN + nt*16 + l16;
        float x = acc[mt][nt][v];
        if (bias) x += bias[col];
        C[(size_t)row * N + col] = x;
      }
}

// ---------------------------------------------------------------------------
// One-time packing kernels (L2-resident, run once per launch)
// ---------------------------------------------------------------------------
// WT[n][k] = bf16(W[k][n])  (weights K x N -> transposed bf16, K fastest)
__global__ void pack_wT(const float* __restrict__ W, unsigned short* __restrict__ WT,
                        int K, int N)
{
  int idx = blockIdx.x * blockDim.x + threadIdx.x;
  if (idx >= K * N) return;
  int k = idx % K;
  int n = idx / K;
  WT[idx] = f2bf(W[(size_t)k * N + n]);
}

__global__ void cast_f32_bf16(const float* __restrict__ in, unsigned short* __restrict__ outp, int n)
{
  int idx = blockIdx.x * blockDim.x + threadIdx.x;
  if (idx < n) outp[idx] = f2bf(in[idx]);
}

// VT[c][t] = bf16(V[t][c]) : (T, NKV*HD) f32 -> (NKV*HD, T) bf16, t fastest
__global__ void cast_vT(const float* __restrict__ V, unsigned short* __restrict__ VT)
{
  int idx = blockIdx.x * blockDim.x + threadIdx.x;
  if (idx >= KVW * T_SEQ) return;
  int t = idx % T_SEQ;
  int c = idx / T_SEQ;
  VT[idx] = f2bf(V[(size_t)t * KVW + c]);
}

// ---------------------------------------------------------------------------
// RoPE (f32 in, bf16 out), layout (T, heads, 128), rotate-half convention.
// ---------------------------------------------------------------------------
__global__ void rope_to_bf16(const float* __restrict__ in, const int* __restrict__ pos,
                             unsigned short* __restrict__ outp, int heads)
{
  int idx = blockIdx.x * blockDim.x + threadIdx.x;
  int total = T_SEQ * heads * (HD/2);
  if (idx >= total) return;
  int j   = idx % (HD/2);
  int rem = idx / (HD/2);
  int hh  = rem % heads;
  int t   = rem / heads;
  float invf = powf(1000000.0f, -(float)j / (float)(HD/2));
  float ang  = (float)pos[t] * invf;
  float s, c;
  sincosf(ang, &s, &c);
  size_t base = (size_t)t * heads * HD + (size_t)hh * HD;
  float x1 = in[base + j];
  float x2 = in[base + j + HD/2];
  outp[base + j]        = f2bf(x1 * c - x2 * s);
  outp[base + j + HD/2] = f2bf(x2 * c + x1 * s);
}

// ---------------------------------------------------------------------------
// Flash attention: block = (128 q-rows, one head); 8 independent waves x 16
// rows. No K/V LDS: K is d-contiguous, V is pre-transposed (d, t) so both
// B-fragments are direct global b128 loads. Only LDS use is the per-wave
// P-tile for the C-layout -> A-layout bf16 conversion (no barriers needed).
// ---------------------------------------------------------------------------
__global__ __launch_bounds__(256)
void flash_attn_bf16(const unsigned short* __restrict__ Qb,
                     const unsigned short* __restrict__ Kb,
                     const unsigned short* __restrict__ Vt,
                     unsigned short* __restrict__ Oa)
{
  __shared__ __align__(16) unsigned short Ps[8][16][128];   // 32 KB, per-wave

  const int tid  = threadIdx.x;
  const int lane = tid & 31;
  const int wave = tid >> 5;
  const int l16  = lane & 15;
  const int hs   = (lane >> 4) & 1;
  const int h    = blockIdx.y;
  const int kvh  = h / GQA;
  const int qbase = blockIdx.x * 128;

  // Q A-fragments, held in VGPRs for the whole kernel
  Frag16 qf[4];
  {
    const unsigned short* qrow = Qb + (size_t)(qbase + wave*16 + l16) * HID + h * HD;
#pragma unroll
    for (int kt = 0; kt < 4; ++kt)
      qf[kt] = load_afrag(qrow + kt*32, hs);
  }

  const unsigned short* kbase = Kb + (size_t)l16 * KVW + kvh * HD;  // + key offsets
  const unsigned short* vbase = Vt + ((size_t)kvh * HD + l16) * T_SEQ;

  f32x8 o[8];
#pragma unroll
  for (int i = 0; i < 8; ++i) o[i] = zero8();
  float mrow[8], lrow[8];
#pragma unroll
  for (int v = 0; v < 8; ++v) { mrow[v] = -1e30f; lrow[v] = 0.f; }

  for (int s0 = 0; s0 < T_SEQ; s0 += 128) {
    // ---- S = Q @ K^T : B-frag column = key (lane), K index = d (contiguous)
    const unsigned short* krow = kbase + (size_t)s0 * KVW;
    f32x8 sfr[8];
#pragma unroll
    for (int nt = 0; nt < 8; ++nt) {
      sfr[nt] = zero8();
#pragma unroll
      for (int kt = 0; kt < 4; ++kt) {
        Frag16 b = load_bfrag(krow + (size_t)nt * 16 * KVW + kt*32, hs);
        sfr[nt] = wmma_bf16(qf[kt].v, b.v, sfr[nt]);
      }
    }

    // ---- online softmax (per-row, 16-lane reductions)
#pragma unroll
    for (int v = 0; v < 8; ++v) {
      float rmax = -1e30f;
#pragma unroll
      for (int nt = 0; nt < 8; ++nt) { sfr[nt][v] *= ATT_SCALE; rmax = fmaxf(rmax, sfr[nt][v]); }
#pragma unroll
      for (int m = 1; m < 16; m <<= 1) rmax = fmaxf(rmax, __shfl_xor(rmax, m, 32));
      float mnew = fmaxf(mrow[v], rmax);
      float sf = __expf(mrow[v] - mnew);
      mrow[v] = mnew;
      float rsum = 0.f;
#pragma unroll
      for (int nt = 0; nt < 8; ++nt) { float p = __expf(sfr[nt][v] - mnew); sfr[nt][v] = p; rsum += p; }
#pragma unroll
      for (int m = 1; m < 16; m <<= 1) rsum += __shfl_xor(rsum, m, 32);
      lrow[v] = lrow[v] * sf + rsum;
#pragma unroll
      for (int nt = 0; nt < 8; ++nt) o[nt][v] *= sf;
    }

    // ---- P: C-layout f32 -> per-wave LDS -> A-layout bf16 (wave-local)
#pragma unroll
    for (int nt = 0; nt < 8; ++nt)
#pragma unroll
      for (int v = 0; v < 8; ++v)
        Ps[wave][v + hs*8][nt*16 + l16] = f2bf(sfr[nt][v]);
    asm volatile("s_wait_dscnt 0" ::: "memory");
    Frag16 pf[4];
#pragma unroll
    for (int kt = 0; kt < 4; ++kt)
      pf[kt] = load_afrag(&Ps[wave][l16][kt*32], hs);

    // ---- O += P @ V : B-frag column = d (lane), K index = key (contiguous in VT)
#pragma unroll
    for (int nt = 0; nt < 8; ++nt) {
      const unsigned short* vrow = vbase + (size_t)nt * 16 * T_SEQ + s0;
#pragma unroll
      for (int kt = 0; kt < 4; ++kt) {
        Frag16 b = load_bfrag(vrow + kt*32, hs);
        o[nt] = wmma_bf16(pf[kt].v, b.v, o[nt]);
      }
    }
  }

  // ---- epilogue: normalize, write bf16 attn
#pragma unroll
  for (int v = 0; v < 8; ++v) {
    float inv = 1.f / lrow[v];
    int row = qbase + wave*16 + v + hs*8;
#pragma unroll
    for (int nt = 0; nt < 8; ++nt) {
      int d = nt*16 + l16;
      Oa[(size_t)row * HID + h*HD + d] = f2bf(o[nt][v] * inv);
    }
  }
}

// ---------------------------------------------------------------------------
extern "C" void kernel_launch(void* const* d_in, const int* in_sizes, int n_in,
                              void* d_out, int out_size, void* d_ws, size_t ws_size,
                              hipStream_t stream)
{
  const int*   positions = (const int*)  d_in[0];
  const float* hidden    = (const float*)d_in[1];
  const float* wq = (const float*)d_in[2];
  const float* bq = (const float*)d_in[3];
  const float* wk = (const float*)d_in[4];
  const float* bk = (const float*)d_in[5];
  const float* wv = (const float*)d_in[6];
  const float* bv = (const float*)d_in[7];
  const float* wo = (const float*)d_in[8];
  float* out = (float*)d_out;
  (void)in_sizes; (void)n_in; (void)out_size; (void)ws_size;

  char* ws = (char*)d_ws;
  size_t off = 0;
  auto carve = [&](size_t bytes) -> char* {
    char* p = ws + off;
    off = (off + bytes + 255) & ~(size_t)255;
    return p;
  };
  unsigned short* hid_bf = (unsigned short*)carve((size_t)T_SEQ * HID * 2);
  unsigned short* wqT    = (unsigned short*)carve((size_t)HID * HID * 2);
  unsigned short* wkT    = (unsigned short*)carve((size_t)HID * KVW * 2);
  unsigned short* wvT    = (unsigned short*)carve((size_t)HID * KVW * 2);
  unsigned short* woT    = (unsigned short*)carve((size_t)HID * HID * 2);
  float*          q_f32  = (float*)         carve((size_t)T_SEQ * HID * 4);
  float*          k_f32  = (float*)         carve((size_t)T_SEQ * KVW * 4);
  float*          v_f32  = (float*)         carve((size_t)T_SEQ * KVW * 4);
  unsigned short* q_bf   = (unsigned short*)carve((size_t)T_SEQ * HID * 2);
  unsigned short* k_bf   = (unsigned short*)carve((size_t)T_SEQ * KVW * 2);
  unsigned short* vt_bf  = (unsigned short*)carve((size_t)KVW * T_SEQ * 2);
  unsigned short* at_bf  = (unsigned short*)carve((size_t)T_SEQ * HID * 2);

  // One-time bf16 packing (activations row-major, weights transposed)
  cast_f32_bf16<<<(T_SEQ*HID + 255)/256, 256, 0, stream>>>(hidden, hid_bf, T_SEQ*HID);
  pack_wT<<<(HID*HID + 255)/256, 256, 0, stream>>>(wq, wqT, HID, HID);
  pack_wT<<<(HID*KVW + 255)/256, 256, 0, stream>>>(wk, wkT, HID, KVW);
  pack_wT<<<(HID*KVW + 255)/256, 256, 0, stream>>>(wv, wvT, HID, KVW);
  pack_wT<<<(HID*HID + 255)/256, 256, 0, stream>>>(wo, woT, HID, HID);

  // QKV projections (register-blocked WMMA TN GEMM, f32 out + bias)
  gemm_tn_bf16<<<dim3(T_SEQ/128, HID/256), 256, 0, stream>>>(hid_bf, wqT, bq, q_f32, HID, HID);
  gemm_tn_bf16<<<dim3(T_SEQ/128, KVW/256), 256, 0, stream>>>(hid_bf, wkT, bk, k_f32, KVW, HID);
  gemm_tn_bf16<<<dim3(T_SEQ/128, KVW/256), 256, 0, stream>>>(hid_bf, wvT, bv, v_f32, KVW, HID);

  // RoPE + bf16 casts (V transposed to (d, t) for contiguous-key fragments)
  rope_to_bf16<<<(T_SEQ*NH *(HD/2) + 255)/256, 256, 0, stream>>>(q_f32, positions, q_bf, NH);
  rope_to_bf16<<<(T_SEQ*NKV*(HD/2) + 255)/256, 256, 0, stream>>>(k_f32, positions, k_bf, NKV);
  cast_vT<<<(KVW*T_SEQ + 255)/256, 256, 0, stream>>>(v_f32, vt_bf);

  // Flash attention (no materialized scores), bf16 attn out
  flash_attn_bf16<<<dim3(T_SEQ/128, NH), 256, 0, stream>>>(q_bf, k_bf, vt_bf, at_bf);

  // Output projection
  gemm_tn_bf16<<<dim3(T_SEQ/128, HID/256), 256, 0, stream>>>(at_bf, woT, nullptr, out, HID, HID);
}
